// InfiniAttention_48052094108063
// MI455X (gfx1250) — matrix operand order repaired and checked by reading, not compiled, not run on hardware
//
#include <hip/hip_runtime.h>

// ---------------------------------------------------------------------------
// InfiniAttention for MI455X (gfx1250, wave32, WMMA + async global->LDS).
//
// Pipeline (all matrix math on v_wmma_f32_16x16x32_bf16, fp32 accumulate):
//   1. convert hidden/weights -> bf16; transpose memory -> memT bf16
//   2. q,k,v = hs @ W^T            (WMMA GEMM, fp32 out)
//   3. sq,sk = elu(q|k)+1 -> bf16 ; vT = transpose(v) -> bf16
//   4. a_mem = sq @ memT^T         (WMMA GEMM)   ; norm = sq . memory_norm
//   5. P = tril(sq @ sk^T) -> bf16 (WMMA GEMM, causal mask in epilogue)
//      (quadratic re-expression of the chunked linear-attention scan:
//       identical algebra, but fully parallel on a WGP-rich chip)
//   6. den = rowsum(P) ; num = P @ vT^T (WMMA GEMM)
//   7. pre = g*mem_out + (1-g)*num/den -> bf16
//   8. out = pre @ w_o^T           (WMMA GEMM, fp32 to d_out)
//
// GEMM staging uses GLOBAL_LOAD_ASYNC_TO_LDS_B128 (ASYNCcnt) with LDS double
// buffering: tile t+1 streams into LDS while tile t feeds the WMMAs.
// ---------------------------------------------------------------------------

typedef unsigned short u16;
typedef __attribute__((ext_vector_type(16))) __bf16 v16bf;
typedef __attribute__((ext_vector_type(8)))  float  v8f;
typedef __attribute__((ext_vector_type(4)))  unsigned int u32x4;

union Frag { v16bf bf; u32x4 q[2]; };

__device__ __forceinline__ u16 f2bf(float x) {
  unsigned u = __float_as_uint(x);
  unsigned r = (u + 0x7FFFu + ((u >> 16) & 1u)) >> 16;   // round-to-nearest-even
  return (u16)r;
}
__device__ __forceinline__ float bf2f(u16 h) {
  return __uint_as_float(((unsigned)h) << 16);
}

// Generic-pointer -> LDS byte offset (addrspacecast folds the round trip).
typedef __attribute__((address_space(3))) const u16 lds_cu16;
__device__ __forceinline__ unsigned lds_addr(const u16* p) {
  return (unsigned)(unsigned long long)(lds_cu16*)p;
}

// One 16B/lane async copy: global -> LDS, tracked by ASYNCcnt (no VGPR data).
__device__ __forceinline__ void async_ld16(unsigned lds, unsigned long long g) {
  asm volatile("global_load_async_to_lds_b128 %0, %1, off"
               :: "v"(lds), "v"(g) : "memory");
}

#define TM 128
#define TN 128
#define TK 32
#define LSTRIDE 40   // 32 halves + 8 pad: keeps b128 alignment, breaks bank conflicts

// ---------------------------------------------------------------------------
// Generic NT GEMM: C[b][m][n] = sum_k A[b][m*lda+k] * B[b][n*ldb+k]
// A,B bf16 (raw u16 bits). Optional fp32 and/or bf16 stores, optional causal
// mask (zero where n > m) applied before both store and bf16 conversion.
// Block: 256 threads = 8 waves; wave w owns a 32x64 region = 2x4 WMMA tiles.
// ---------------------------------------------------------------------------
__global__ __launch_bounds__(256, 2)
void wmma_gemm_nt_kernel(const u16* __restrict__ A, int lda, long long sA,
                         const u16* __restrict__ B, int ldb, long long sB,
                         float* __restrict__ Cf, u16* __restrict__ Cb,
                         int ldc, long long sC, int K, int causal)
{
  __shared__ u16 As[2][TM][LSTRIDE];
  __shared__ u16 Bs[2][TN][LSTRIDE];

  const int bz = blockIdx.z;
  const u16* Ab = A + (size_t)bz * (size_t)sA;
  const u16* Bb = B + (size_t)bz * (size_t)sB;
  const int m0 = blockIdx.y * TM;
  const int n0 = blockIdx.x * TN;

  const int tid = threadIdx.x;
  const int w  = tid >> 5;
  const int l  = tid & 31;
  const int mr = (w >> 1) * 32;      // wave row base within tile (0..96)
  const int nc = (w & 1) * 64;       // wave col base within tile (0/64)
  const int lr = l & 15;
  const int hi = l >> 4;             // lane half select

  v8f acc[2][4];
  for (int i = 0; i < 2; ++i)
    for (int j = 0; j < 4; ++j)
      for (int e = 0; e < 8; ++e) acc[i][j][e] = 0.0f;

  const int srow = tid >> 1;         // staging row 0..127
  const int sseg = (tid & 1) * 16;   // staging half-offset (halves)

  const u16* gaBase = Ab + (size_t)(m0 + srow) * lda + sseg;
  const u16* gbBase = Bb + (size_t)(n0 + srow) * ldb + sseg;
  unsigned lA[2], lB[2];
  lA[0] = lds_addr(&As[0][srow][sseg]);
  lA[1] = lds_addr(&As[1][srow][sseg]);
  lB[0] = lds_addr(&Bs[0][srow][sseg]);
  lB[1] = lds_addr(&Bs[1][srow][sseg]);

  const int nt = K / TK;

  // Stage tile t into LDS buffer `buf`: 4 async b128 per thread (A lo/hi, B
  // lo/hi). Same +16B delta on global and LDS sides within a thread.
  auto stage = [&](int t, int buf) {
    const u16* ga = gaBase + t * TK;
    const u16* gb = gbBase + t * TK;
    async_ld16(lA[buf],      (unsigned long long)ga);
    async_ld16(lA[buf] + 16, (unsigned long long)(ga + 8));
    async_ld16(lB[buf],      (unsigned long long)gb);
    async_ld16(lB[buf] + 16, (unsigned long long)(gb + 8));
    if (t + 1 < nt) {                      // warm L2 one more tile ahead
      __builtin_prefetch(ga + TK, 0, 1);
      __builtin_prefetch(gb + TK, 0, 1);
    }
  };

  stage(0, 0);
  for (int t = 0; t < nt; ++t) {
    const int cur = t & 1;
    if (t + 1 < nt) {
      stage(t + 1, cur ^ 1);               // overlap next tile's DMA
      asm volatile("s_wait_asynccnt %0" :: "i"(4) : "memory");  // tile t done
    } else {
      asm volatile("s_wait_asynccnt %0" :: "i"(0) : "memory");
    }
    __syncthreads();                       // all waves' tile-t data visible

    // A fragments: lane half 0 -> K {0..7,16..23}; half 1 -> K {8..15,24..31}
    Frag af[2], bfr[4];
    const int abase = hi ? 8 : 0;
    for (int rt = 0; rt < 2; ++rt) {
      const int row = mr + rt * 16 + lr;
      af[rt].q[0] = *(const u32x4*)&As[cur][row][abase];
      af[rt].q[1] = *(const u32x4*)&As[cur][row][abase + 16];
    }
    // B fragments: lane half 0 -> K {0..7,16..23}; half 1 -> K {8..15,24..31}
    const int kb = hi ? 8 : 0;
    for (int ct = 0; ct < 4; ++ct) {
      const int col = nc + ct * 16 + lr;
      bfr[ct].q[0] = *(const u32x4*)&Bs[cur][col][kb];
      bfr[ct].q[1] = *(const u32x4*)&Bs[cur][col][kb + 16];
    }
    for (int rt = 0; rt < 2; ++rt)
      for (int ct = 0; ct < 4; ++ct)
        acc[rt][ct] = __builtin_amdgcn_wmma_f32_16x16x32_bf16(
            false, af[rt].bf, false, bfr[ct].bf,
            (short)0, acc[rt][ct], false, false);

    __syncthreads();                       // readers done before buf reuse
  }

  // Epilogue. C layout: vgpr r, lanes 0-15 -> (M=r, N=lane); 16-31 -> (M=8+r)
  for (int rt = 0; rt < 2; ++rt)
    for (int ct = 0; ct < 4; ++ct)
      for (int r = 0; r < 8; ++r) {
        const int m = m0 + mr + rt * 16 + r + hi * 8;
        const int n = n0 + nc + ct * 16 + lr;
        float v = acc[rt][ct][r];
        if (causal && n > m) v = 0.0f;
        const size_t idx = (size_t)bz * (size_t)sC + (size_t)m * ldc + n;
        if (Cf) Cf[idx] = v;
        if (Cb) Cb[idx] = f2bf(v);
      }
}

// ---------------------------------------------------------------------------
// Elementwise / helper kernels
// ---------------------------------------------------------------------------
__global__ void cvt_bf16_kernel(const float* __restrict__ in,
                                u16* __restrict__ out, long long n) {
  long long i = (long long)blockIdx.x * blockDim.x + threadIdx.x;
  if (i < n) out[i] = f2bf(in[i]);
}

__global__ void elu1_bf16_kernel(const float* __restrict__ in,
                                 u16* __restrict__ out, long long n) {
  long long i = (long long)blockIdx.x * blockDim.x + threadIdx.x;
  if (i < n) {
    float x = in[i];
    out[i] = f2bf(x > 0.0f ? x + 1.0f : __expf(x));   // elu(x)+1
  }
}

// in: [batch][M][N] f32 -> out: [batch][N][M] bf16
__global__ __launch_bounds__(256)
void transpose_bf16_kernel(const float* __restrict__ in, u16* __restrict__ out,
                           int M, int N, long long sIn, long long sOut) {
  __shared__ float tile[32][33];
  const int b  = blockIdx.z;
  const float* ip = in + (size_t)b * (size_t)sIn;
  u16* op = out + (size_t)b * (size_t)sOut;
  const int m0 = blockIdx.y * 32, n0 = blockIdx.x * 32;
  const int tx = threadIdx.x & 31, ty = threadIdx.x >> 5;
  for (int i = 0; i < 32; i += 8)
    tile[ty + i][tx] = ip[(size_t)(m0 + ty + i) * N + n0 + tx];
  __syncthreads();
  for (int i = 0; i < 32; i += 8)
    op[(size_t)(n0 + ty + i) * M + m0 + tx] = f2bf(tile[tx][ty + i]);
}

// norm[r] = sum_d sq[r][d] * memnorm[d]
__global__ void normdot_kernel(const u16* __restrict__ sq,
                               const float* __restrict__ mn,
                               float* __restrict__ out, int rows, int D) {
  int r = blockIdx.x * blockDim.x + threadIdx.x;
  if (r >= rows) return;
  const u16* p = sq + (size_t)r * D;
  float s = 0.0f;
  for (int d = 0; d < D; ++d) s += bf2f(p[d]) * mn[d];
  out[r] = s;
}

// den[b*S+t] = sum_j P[b][t][j]   (P already causally masked)
__global__ __launch_bounds__(128)
void rowsum_kernel(const u16* __restrict__ P, float* __restrict__ den, int S) {
  const int row = blockIdx.x;           // b*S + t
  const int b = row / S, t = row % S;
  const u16* pr = P + (size_t)b * S * S + (size_t)t * S;
  float s = 0.0f;
  for (int j = threadIdx.x; j < S; j += 128) s += bf2f(pr[j]);
  __shared__ float red[128];
  red[threadIdx.x] = s;
  __syncthreads();
  for (int st = 64; st > 0; st >>= 1) {
    if (threadIdx.x < st) red[threadIdx.x] += red[threadIdx.x + st];
    __syncthreads();
  }
  if (threadIdx.x == 0) den[row] = red[0];
}

// scal[0] = sigmoid(gate), scal[1] = (sum(memory_norm) >= eps)
__global__ __launch_bounds__(256)
void scalars_kernel(const float* __restrict__ gate,
                    const float* __restrict__ mn, float* __restrict__ scal,
                    int D) {
  __shared__ float red[256];
  float s = 0.0f;
  for (int i = threadIdx.x; i < D; i += 256) s += mn[i];
  red[threadIdx.x] = s;
  __syncthreads();
  for (int st = 128; st > 0; st >>= 1) {
    if (threadIdx.x < st) red[threadIdx.x] += red[threadIdx.x + st];
    __syncthreads();
  }
  if (threadIdx.x == 0) {
    scal[0] = 1.0f / (1.0f + __expf(-gate[0]));
    scal[1] = (red[0] >= 1e-6f) ? 1.0f : 0.0f;
  }
}

// pre = bf16( g*act*(amem/max(norm,eps)) + (1-g)*(num/max(den,eps)) )
__global__ void combine_kernel(const float* __restrict__ num,
                               const float* __restrict__ den,
                               const float* __restrict__ amem,
                               const float* __restrict__ nrm,
                               const float* __restrict__ scal,
                               u16* __restrict__ pre, long long n, int D) {
  long long i = (long long)blockIdx.x * blockDim.x + threadIdx.x;
  if (i >= n) return;
  const long long t = i / D;
  const float g = scal[0], act = scal[1];
  const float local = num[i] / fmaxf(den[t], 1e-6f);
  const float mem   = act * (amem[i] / fmaxf(nrm[t], 1e-6f));
  pre[i] = f2bf(g * mem + (1.0f - g) * local);
}

// ---------------------------------------------------------------------------
// Host-side orchestration
// ---------------------------------------------------------------------------
extern "C" void kernel_launch(void* const* d_in, const int* in_sizes, int n_in,
                              void* d_out, int out_size, void* d_ws, size_t ws_size,
                              hipStream_t stream) {
  constexpr int Bn = 2, S = 2048, D = 1024;
  const size_t nTok = (size_t)Bn * S;       // 4096 rows
  const size_t nHid = nTok * D;             // B*S*D
  const size_t nW   = (size_t)D * D;
  const size_t nP   = (size_t)Bn * S * S;

  const float* hs      = (const float*)d_in[0];
  const float* w_q     = (const float*)d_in[1];
  const float* w_k     = (const float*)d_in[2];
  const float* w_v     = (const float*)d_in[3];
  const float* w_o     = (const float*)d_in[4];
  const float* gate    = (const float*)d_in[5];
  const float* memory  = (const float*)d_in[6];
  const float* memnorm = (const float*)d_in[7];
  float* out = (float*)d_out;

  char* ws = (char*)d_ws;
  size_t off = 0;
  auto alloc = [&](size_t bytes) -> void* {
    void* p = ws + off;
    off = (off + bytes + 255) & ~(size_t)255;
    return p;
  };
  u16*   hs_bf   = (u16*)alloc(nHid * 2);
  u16*   wq_bf   = (u16*)alloc(nW * 2);
  u16*   wk_bf   = (u16*)alloc(nW * 2);
  u16*   wv_bf   = (u16*)alloc(nW * 2);
  u16*   wo_bf   = (u16*)alloc(nW * 2);
  u16*   memT_bf = (u16*)alloc(nW * 2);
  float* q_f     = (float*)alloc(nHid * 4);
  float* k_f     = (float*)alloc(nHid * 4);
  float* v_f     = (float*)alloc(nHid * 4);
  u16*   sq_bf   = (u16*)alloc(nHid * 2);
  u16*   sk_bf   = (u16*)alloc(nHid * 2);
  u16*   vT_bf   = (u16*)alloc(nHid * 2);
  float* amem_f  = (float*)alloc(nHid * 4);
  float* norm_f  = (float*)alloc(nTok * 4);
  u16*   P_bf    = (u16*)alloc(nP * 2);
  float* den_f   = (float*)alloc(nTok * 4);
  float* num_f   = (float*)alloc(nHid * 4);
  u16*   pre_bf  = (u16*)alloc(nHid * 2);
  float* scal    = (float*)alloc(256);
  (void)ws_size; (void)n_in; (void)in_sizes; (void)out_size;

  const int T = 256;
  auto blocks = [&](long long n) { return (unsigned)((n + T - 1) / T); };

  // 1) conversions
  cvt_bf16_kernel<<<blocks(nHid), T, 0, stream>>>(hs, hs_bf, (long long)nHid);
  cvt_bf16_kernel<<<blocks(nW), T, 0, stream>>>(w_q, wq_bf, (long long)nW);
  cvt_bf16_kernel<<<blocks(nW), T, 0, stream>>>(w_k, wk_bf, (long long)nW);
  cvt_bf16_kernel<<<blocks(nW), T, 0, stream>>>(w_v, wv_bf, (long long)nW);
  cvt_bf16_kernel<<<blocks(nW), T, 0, stream>>>(w_o, wo_bf, (long long)nW);
  transpose_bf16_kernel<<<dim3(D / 32, D / 32, 1), T, 0, stream>>>(
      memory, memT_bf, D, D, 0, 0);

  // 2) projections: [4096 x 1024] = hs @ W^T, K=1024
  dim3 gproj(D / TN, (unsigned)(nTok / TM), 1);
  wmma_gemm_nt_kernel<<<gproj, T, 0, stream>>>(hs_bf, D, 0, wq_bf, D, 0,
                                               q_f, nullptr, D, 0, D, 0);
  wmma_gemm_nt_kernel<<<gproj, T, 0, stream>>>(hs_bf, D, 0, wk_bf, D, 0,
                                               k_f, nullptr, D, 0, D, 0);
  wmma_gemm_nt_kernel<<<gproj, T, 0, stream>>>(hs_bf, D, 0, wv_bf, D, 0,
                                               v_f, nullptr, D, 0, D, 0);

  // 3) feature maps + v transpose
  elu1_bf16_kernel<<<blocks(nHid), T, 0, stream>>>(q_f, sq_bf, (long long)nHid);
  elu1_bf16_kernel<<<blocks(nHid), T, 0, stream>>>(k_f, sk_bf, (long long)nHid);
  transpose_bf16_kernel<<<dim3(D / 32, S / 32, Bn), T, 0, stream>>>(
      v_f, vT_bf, S, D, (long long)S * D, (long long)D * S);

  // 4) compressive-memory retrieval
  wmma_gemm_nt_kernel<<<gproj, T, 0, stream>>>(sq_bf, D, 0, memT_bf, D, 0,
                                               amem_f, nullptr, D, 0, D, 0);
  normdot_kernel<<<blocks(nTok), T, 0, stream>>>(sq_bf, memnorm, norm_f,
                                                 (int)nTok, D);

  // 5) P = tril(sq @ sk^T) per batch -> bf16  (quadratic == scan algebra)
  dim3 gP(S / TN, S / TM, Bn);
  wmma_gemm_nt_kernel<<<gP, T, 0, stream>>>(
      sq_bf, D, (long long)S * D, sk_bf, D, (long long)S * D,
      nullptr, P_bf, S, (long long)S * S, D, 1);

  // 6) den = rowsum(P); num = P @ v  (B operand = vT, NT form, K=S)
  rowsum_kernel<<<(unsigned)nTok, 128, 0, stream>>>(P_bf, den_f, S);
  dim3 gnum(D / TN, S / TM, Bn);
  wmma_gemm_nt_kernel<<<gnum, T, 0, stream>>>(
      P_bf, S, (long long)S * S, vT_bf, S, (long long)D * S,
      num_f, nullptr, D, (long long)S * D, S, 0);

  // 7) gated combine -> bf16
  scalars_kernel<<<1, T, 0, stream>>>(gate, memnorm, scal, D);
  combine_kernel<<<blocks(nHid), T, 0, stream>>>(num_f, den_f, amem_f, norm_f,
                                                 scal, pre_bf, (long long)nHid, D);

  // 8) output projection -> d_out (fp32)
  wmma_gemm_nt_kernel<<<gproj, T, 0, stream>>>(pre_bf, D, 0, wo_bf, D, 0,
                                               out, nullptr, D, 0, D, 0);
}
